// TemporalAttention_3152505995658
// MI455X (gfx1250) — matrix operand order repaired and checked by reading, not compiled, not run on hardware
//
#include <hip/hip_runtime.h>
#include <stdint.h>

#define B_DIM 128
#define S_DIM 512
#define H_DIM 512

typedef __attribute__((ext_vector_type(16))) _Float16 v16h;
typedef __attribute__((ext_vector_type(8)))  _Float16 h8;
typedef __attribute__((ext_vector_type(4)))  _Float16 h4;
typedef __attribute__((ext_vector_type(8)))  float    v8f;

// ---------------------------------------------------------------------------
// Branch-free tanh: native V_TANH_F32 on gfx1250 (confirmed in round-2 asm).
// ---------------------------------------------------------------------------
__device__ __forceinline__ float fast_tanh(float v) {
#if __has_builtin(__builtin_amdgcn_tanhf)
    return __builtin_amdgcn_tanhf(v);
#elif __has_builtin(__builtin_amdgcn_tanh_f32)
    return __builtin_amdgcn_tanh_f32(v);
#else
    const float t = __expf(2.0f * v);
    return 1.0f - 2.0f * __builtin_amdgcn_rcpf(t + 1.0f);
#endif
}

// ---------------------------------------------------------------------------
// Kernel 1: convert U1a (U1[:, :H]) and U2 to f16 in workspace.
// ---------------------------------------------------------------------------
__global__ __launch_bounds__(256) void cvt_weights_kernel(
    const float* __restrict__ U1, const float* __restrict__ U2,
    _Float16* __restrict__ wa, _Float16* __restrict__ w2)
{
    const int k = blockIdx.x;                    // 0..511
    for (int hh = threadIdx.x; hh < H_DIM; hh += blockDim.x) {
        wa[(size_t)k * H_DIM + hh] = (_Float16)U1[(size_t)k * (2 * H_DIM) + hh];
        w2[(size_t)k * H_DIM + hh] = (_Float16)U2[(size_t)k * H_DIM + hh];
    }
}

// ---------------------------------------------------------------------------
// Kernel 2: bias[b,k] = sum_h h[b, S-1, h] * U1[k, H + h]   (f32, tiny GEMV)
// ---------------------------------------------------------------------------
__global__ __launch_bounds__(256) void bias_kernel(
    const float* __restrict__ h, const float* __restrict__ U1,
    float* __restrict__ bias)
{
    __shared__ float hl[H_DIM];
    const int b = blockIdx.x;
    const float* hlast = h + ((size_t)b * S_DIM + (S_DIM - 1)) * H_DIM;
    for (int i = threadIdx.x; i < H_DIM; i += blockDim.x) hl[i] = hlast[i];
    __syncthreads();

    const int k = blockIdx.y * blockDim.x + threadIdx.x;     // 0..511
    const float* w = U1 + (size_t)k * (2 * H_DIM) + H_DIM;   // U1b row k
    float acc = 0.0f;
    #pragma unroll 8
    for (int i = 0; i < H_DIM; ++i) acc = fmaf(hl[i], w[i], acc);
    bias[(size_t)b * H_DIM + k] = acc;
}

// ---------------------------------------------------------------------------
// Kernel 3: fused  e[b,s] = sum_k Ve[k]*tanh( h.U1a^T + x.U2^T + bias[b,k] )
// One wave per 16 rows. A (h,x rows) staged as f16 in LDS, read ONCE per
// k-step and reused across NB=4 weight tiles (4 accumulators) -> 0.5 ds_load
// per WMMA instead of 2, matching WGP LDS bandwidth to the matrix pipe.
// ---------------------------------------------------------------------------
#define ROWS 16
#define KPAD 520   // 512+8 pad: lane m -> dword offset 260*m -> banks 4m..4m+3, conflict-free
#define NB   4     // n-tiles register-blocked per A-fragment load

__global__ __launch_bounds__(32) void attn_e_kernel(
    const float* __restrict__ h, const float* __restrict__ x,
    const _Float16* __restrict__ wa, const _Float16* __restrict__ w2,
    const float* __restrict__ bias, const float* __restrict__ ve,
    float* __restrict__ e)
{
    __shared__ _Float16 hA[ROWS][KPAD];
    __shared__ _Float16 xA[ROWS][KPAD];

    const int lane = threadIdx.x;                         // 0..31
    const size_t rowBase = (size_t)blockIdx.x * ROWS;     // flattened (b*s) row
    const int b     = (int)(rowBase / S_DIM);
    const int sBase = (int)(rowBase % S_DIM);

    // ---- stage 16 rows of h and x, converted to f16, into LDS ----
    const float* hsrc = h + rowBase * H_DIM;
    const float* xsrc = x + rowBase * H_DIM;
    for (int idx = lane; idx < ROWS * (H_DIM / 4); idx += 32) {
        const int r  = idx >> 7;             // / 128
        const int c4 = (idx & 127) << 2;     // *4
        const float4 hv = *(const float4*)(hsrc + (size_t)r * H_DIM + c4);
        const float4 xv = *(const float4*)(xsrc + (size_t)r * H_DIM + c4);
        h4 hp = { (_Float16)hv.x, (_Float16)hv.y, (_Float16)hv.z, (_Float16)hv.w };
        h4 xp = { (_Float16)xv.x, (_Float16)xv.y, (_Float16)xv.z, (_Float16)xv.w };
        *(h4*)&hA[r][c4] = hp;
        *(h4*)&xA[r][c4] = xp;
    }
    __syncthreads();

    const int lm    = lane & 15;       // A: row within tile / B: column / C: N
    const int hi    = lane >> 4;
    const int koffA = hi * 8;          // A layout: K chunks [koffA,+8) and [koffA+16,+24)
    const int koffB = hi * 8;          // B layout: K chunks [koffB,+8) and [koffB+16,+24), same as A

    float ev[8] = {0.f, 0.f, 0.f, 0.f, 0.f, 0.f, 0.f, 0.f};
    const float* biasRow = bias + (size_t)b * H_DIM;
    const v8f vzero = {0.f, 0.f, 0.f, 0.f, 0.f, 0.f, 0.f, 0.f};

    union Frag { h8 p[2]; v16h v; };

    for (int ntb = 0; ntb < 32 / NB; ++ntb) {
        const int nt0 = ntb * NB;
        const _Float16* pwa[NB];
        const _Float16* pw2[NB];
        #pragma unroll
        for (int j = 0; j < NB; ++j) {
            const int wrow = (nt0 + j) * 16 + lm;            // weight row = output col n
            pwa[j] = wa + (size_t)wrow * H_DIM + koffB;
            pw2[j] = w2 + (size_t)wrow * H_DIM + koffB;
        }
        if (ntb + 1 < 32 / NB) {                             // prefetch next weight block
            __builtin_prefetch(pwa[0] + (size_t)NB * 16 * H_DIM, 0, 3);
            __builtin_prefetch(pw2[0] + (size_t)NB * 16 * H_DIM, 0, 3);
        }

        v8f acc[NB];
        #pragma unroll
        for (int j = 0; j < NB; ++j) acc[j] = vzero;

        #pragma unroll 2
        for (int ks = 0; ks < 16; ++ks) {
            const int k0 = ks * 32;
            Frag ah, ax;                                     // A frags: loaded once, used NB times
            ah.p[0] = *(const h8*)&hA[lm][k0 + koffA];
            ah.p[1] = *(const h8*)&hA[lm][k0 + koffA + 16];
            ax.p[0] = *(const h8*)&xA[lm][k0 + koffA];
            ax.p[1] = *(const h8*)&xA[lm][k0 + koffA + 16];
            #pragma unroll
            for (int j = 0; j < NB; ++j) {
                Frag bm;
                bm.p[0] = *(const h8*)(pwa[j] + k0);         // h * U1a^T
                bm.p[1] = *(const h8*)(pwa[j] + k0 + 16);
                acc[j] = __builtin_amdgcn_wmma_f32_16x16x32_f16(false, ah.v, false, bm.v,
                                                                (short)0, acc[j], false, false);
                bm.p[0] = *(const h8*)(pw2[j] + k0);         // x * U2^T
                bm.p[1] = *(const h8*)(pw2[j] + k0 + 16);
                acc[j] = __builtin_amdgcn_wmma_f32_16x16x32_f16(false, ax.v, false, bm.v,
                                                                (short)0, acc[j], false, false);
            }
        }

        // epilogue: tanh + Ve dot, fused in-register (native v_tanh_f32)
        #pragma unroll
        for (int j = 0; j < NB; ++j) {
            const int   n  = (nt0 + j) * 16 + lm;
            const float bn = biasRow[n];
            const float vn = ve[n];
            #pragma unroll
            for (int i = 0; i < 8; ++i)
                ev[i] += vn * fast_tanh(acc[j][i] + bn);
        }
    }

    // reduce over the 16 column-lanes within each half-wave
    #pragma unroll
    for (int i = 0; i < 8; ++i) {
        float v = ev[i];
        v += __shfl_xor(v, 1, 32);
        v += __shfl_xor(v, 2, 32);
        v += __shfl_xor(v, 4, 32);
        v += __shfl_xor(v, 8, 32);
        ev[i] = v;
    }
    // C layout: lanes 0-15 hold rows M=0..7 (VGPR i -> M=i); lanes 16-31 hold M=8..15
    if (lm == 0) {
        float* eo = e + (size_t)b * S_DIM + sBase + hi * 8;
        #pragma unroll
        for (int i = 0; i < 8; ++i) eo[i] = ev[i];
    }
}

// ---------------------------------------------------------------------------
// Kernel 4: per-batch softmax over s, then context[b,h] = sum_s alpha*h
// ---------------------------------------------------------------------------
__global__ __launch_bounds__(256) void softmax_ctx_kernel(
    const float* __restrict__ h, const float* __restrict__ e,
    float* __restrict__ out)
{
    __shared__ float alpha[S_DIM];
    __shared__ float red[256];
    const int b = blockIdx.x;
    const int t = threadIdx.x;

    const float* eb = e + (size_t)b * S_DIM;
    const float e0 = eb[t], e1 = eb[t + 256];

    red[t] = fmaxf(e0, e1);
    __syncthreads();
    for (int off = 128; off > 0; off >>= 1) {
        if (t < off) red[t] = fmaxf(red[t], red[t + off]);
        __syncthreads();
    }
    const float M = red[0];
    __syncthreads();

    const float a0 = __expf(e0 - M), a1 = __expf(e1 - M);
    red[t] = a0 + a1;
    __syncthreads();
    for (int off = 128; off > 0; off >>= 1) {
        if (t < off) red[t] += red[t + off];
        __syncthreads();
    }
    const float inv = 1.0f / red[0];
    __syncthreads();
    alpha[t]       = a0 * inv;
    alpha[t + 256] = a1 * inv;
    __syncthreads();

    const float* hb = h + (size_t)b * S_DIM * H_DIM;
    float c0 = 0.f, c1 = 0.f;
    #pragma unroll 4
    for (int s = 0; s < S_DIM; ++s) {
        const float a = alpha[s];
        c0 = fmaf(a, hb[(size_t)s * H_DIM + t], c0);
        c1 = fmaf(a, hb[(size_t)s * H_DIM + t + 256], c1);
    }
    out[(size_t)b * H_DIM + t]       = c0;
    out[(size_t)b * H_DIM + t + 256] = c1;
}

// ---------------------------------------------------------------------------
extern "C" void kernel_launch(void* const* d_in, const int* in_sizes, int n_in,
                              void* d_out, int out_size, void* d_ws, size_t ws_size,
                              hipStream_t stream)
{
    const float* h  = (const float*)d_in[0];   // (B,S,H)
    const float* x  = (const float*)d_in[1];   // (B,S,H)
    const float* Ve = (const float*)d_in[2];   // (1,H)
    const float* U1 = (const float*)d_in[3];   // (H,2H)
    const float* U2 = (const float*)d_in[4];   // (H,H)
    float* out = (float*)d_out;                // (B,H)

    char* ws = (char*)d_ws;
    _Float16* wa  = (_Float16*)ws;                                       // 512 KB
    _Float16* w2  = (_Float16*)(ws + (size_t)H_DIM * H_DIM * 2);         // 512 KB
    float*    bia = (float*)   (ws + (size_t)H_DIM * H_DIM * 4);         // 256 KB
    float*    e   = (float*)   (ws + (size_t)H_DIM * H_DIM * 4
                                   + (size_t)B_DIM * H_DIM * 4);         // 256 KB

    cvt_weights_kernel<<<H_DIM, 256, 0, stream>>>(U1, U2, wa, w2);
    bias_kernel<<<dim3(B_DIM, H_DIM / 256), 256, 0, stream>>>(h, U1, bia);
    attn_e_kernel<<<(B_DIM * S_DIM) / ROWS, 32, 0, stream>>>(h, x, wa, w2, bia, Ve, e);
    softmax_ctx_kernel<<<B_DIM, 256, 0, stream>>>(h, e, out);
}
